// EncodeBlock_40372692583277
// MI455X (gfx1250) — compile-verified, matrix operand order checked by reading, not run on hardware
//
#include <hip/hip_runtime.h>

// ---------------------------------------------------------------------------
// Types for CDNA5 WMMA (wave32, v_wmma_f32_16x16x32_bf16)
// ---------------------------------------------------------------------------
typedef __bf16 bf16_t;
typedef __attribute__((ext_vector_type(16))) __bf16 v16bf;
typedef __attribute__((ext_vector_type(8)))  float  v8f;
typedef __attribute__((address_space(3))) void lds_void;

constexpr int B = 8, S = 512, D = 1024, H = 8, HD = 128, HALF = 64;
constexpr int BS = B * S;                 // 4096 rows
constexpr float EPS = 1e-6f;

__device__ __forceinline__ v8f vzero8() {
  v8f z;
  for (int i = 0; i < 8; ++i) z[i] = 0.f;
  return z;
}

// Load one 16x32 bf16 fragment (A or B operand) per the CDNA5 16-bit layout:
// lane<16 holds K {0..7,16..23}, lane>=16 holds K {8..15,24..31}; caller has
// already folded row*stride + (lane>>4)*8 into p. Two b128 loads.
__device__ __forceinline__ v16bf ld_frag(const bf16_t* p) {
  union { uint4 u[2]; v16bf v; } U;
  U.u[0] = *(const uint4*)(p);
  U.u[1] = *(const uint4*)(p + 16);
  return U.v;
}

__device__ __forceinline__ v8f wmma_bf16(v16bf a, v16bf b, v8f c) {
  return __builtin_amdgcn_wmma_f32_16x16x32_bf16(
      /*neg_a=*/false, a, /*neg_b=*/false, b,
      /*c_mod=*/(short)0, c, /*reuse_a=*/false, /*reuse_b=*/false);
}

// CDNA5 async global->LDS copy (ASYNCcnt-tracked DMA path, ISA 15.18.3 op 98).
// Bypasses VGPR staging entirely: LDS byte address in one VGPR, 64-bit global
// address in a VGPR pair.
__device__ __forceinline__ void async_copy_b128(const void* gsrc, void* ldst) {
  unsigned loff = (unsigned)(unsigned long long)(lds_void*)ldst;
  unsigned long long ga = (unsigned long long)gsrc;
  asm volatile("global_load_async_to_lds_b128 %0, %1, off"
               :: "v"(loff), "v"(ga)
               : "memory");
}
__device__ __forceinline__ void wait_async0() {
  asm volatile("s_wait_asynccnt 0x0" ::: "memory");
}

__device__ __forceinline__ float siluf(float v) {
  return v / (1.f + __expf(-v));
}

// ---------------------------------------------------------------------------
// Prep kernels
// ---------------------------------------------------------------------------

// wT[n*D + k] = (bf16) w[k*D + n]   (weights pre-transposed for B fragments)
__global__ __launch_bounds__(256) void transpose_w_kernel(const float* w, bf16_t* wT) {
  int idx = blockIdx.x * 256 + threadIdx.x;          // idx = k*D + n
  int k = idx / D, n = idx % D;
  wT[(size_t)n * D + k] = (bf16_t)w[idx];
}

// hb[bh][e][d] = (bf16) hstate[bh][d][e]
__global__ __launch_bounds__(256) void hb_transpose_kernel(const float* hs, bf16_t* hb) {
  int idx = blockIdx.x * 256 + threadIdx.x;          // bh*HD*HD + d*HD + e
  int e = idx % HD, d = (idx / HD) % HD, bh = idx / (HD * HD);
  hb[((size_t)bh * HD + e) * HD + d] = (bf16_t)hs[idx];
}

// Exclusive cumsum of dones per batch (dcx = cumsum(done) - done)
__global__ void dcx_kernel(const unsigned char* dones, int* dcx) {
  int b = blockIdx.x;
  int run = 0;
  for (int s = 0; s < S; ++s) {
    int dv = dones[b * S + s] ? 1 : 0;
    dcx[b * S + s] = run;
    run += dv;
  }
}

// RMSNorm over D; writes f32 (optional) and bf16 copies.
__global__ __launch_bounds__(256) void rmsnorm_kernel(const float* x, const float* wt,
                                                      float* of, bf16_t* ob) {
  __shared__ float red[256];
  const int row = blockIdx.x, tid = threadIdx.x;
  const float* xr = x + (size_t)row * D;
  float ss = 0.f;
  for (int i = tid; i < D; i += 256) { float v = xr[i]; ss += v * v; }
  red[tid] = ss;
  __syncthreads();
  for (int s = 128; s > 0; s >>= 1) {
    if (tid < s) red[tid] += red[tid + s];
    __syncthreads();
  }
  const float inv = rsqrtf(red[0] / (float)D + EPS);
  for (int i = tid; i < D; i += 256) {
    float v = xr[i] * inv * wt[i];
    if (of) of[(size_t)row * D + i] = v;
    ob[(size_t)row * D + i] = (bf16_t)v;
  }
}

// Rotary + scale + cast to bf16.  idx enumerates (b,s,h,j<64).
__global__ void rotary_kernel(const float* src, bf16_t* dst, const int* step, float scale) {
  int idx = blockIdx.x * blockDim.x + threadIdx.x;
  int j = idx & (HALF - 1);
  int h = (idx >> 6) & (H - 1);
  int s = (idx >> 9) & (S - 1);
  int b = idx >> 18;
  float t = (float)step[b * S + s];
  float fr = __powf(10000.f, -(float)j / (float)HALF);
  float sn, cs;
  __sincosf(t * fr, &sn, &cs);
  size_t base = ((size_t)(b * S + s)) * D + h * HD;
  float x1 = src[base + j], x2 = src[base + HALF + j];
  dst[base + j]        = (bf16_t)((x1 * cs - x2 * sn) * scale);
  dst[base + HALF + j] = (bf16_t)((x1 * sn + x2 * cs) * scale);
}

__global__ void cvt_f2b_kernel(const float* src, bf16_t* dst, int n) {
  int i = blockIdx.x * blockDim.x + threadIdx.x;
  if (i < n) dst[i] = (bf16_t)src[i];
}

__global__ void silu_mul_kernel(const float* a, const float* m, bf16_t* o, int n) {
  int i = blockIdx.x * blockDim.x + threadIdx.x;
  if (i < n) o[i] = (bf16_t)(siluf(a[i]) * m[i]);
}

// Group-norm per (b,s,h) over HD=128 + SiLU gating; one wave per group.
__global__ __launch_bounds__(256) void gn_gate_kernel(const float* ret, const float* g,
                                                      const float* sc, const float* bi,
                                                      bf16_t* outg) {
  int grp = blockIdx.x * 8 + (threadIdx.x >> 5);     // grp = (b*S + s)*H + h
  int lane = threadIdx.x & 31;
  int h = grp % H, s = (grp / H) % S, b = grp / (H * S);
  size_t base = ((size_t)(b * S + s)) * D + h * HD;
  float r[4], s1 = 0.f, s2 = 0.f;
  for (int i = 0; i < 4; ++i) {
    r[i] = ret[base + lane + i * 32];
    s1 += r[i];
    s2 += r[i] * r[i];
  }
  for (int m = 16; m >= 1; m >>= 1) {
    s1 += __shfl_xor(s1, m, 32);
    s2 += __shfl_xor(s2, m, 32);
  }
  float mu = s1 / (float)HD;
  float var = s2 / (float)HD - mu * mu;
  float rs = rsqrtf(var + EPS);
  for (int i = 0; i < 4; ++i) {
    int d = lane + i * 32;
    float rn = (r[i] - mu) * rs * sc[h * HD + d] + bi[h * HD + d];
    float gv = g[base + d];
    outg[base + d] = (bf16_t)(siluf(gv) * rn);
  }
}

// ---------------------------------------------------------------------------
// Generic bf16 WMMA GEMM: C[M][N] = A[M][K] (row-major bf16) x Bt[N][K] (bf16)
// optional f32 residual fused in the epilogue.
// 128x128 block tile, 8 waves, each wave: 16 rows x 128 cols (8 accumulators).
// Tile staging uses GLOBAL_LOAD_ASYNC_TO_LDS_B128 (no VGPR round-trip); all 8
// B fragments are preloaded so the 8 WMMAs issue back-to-back.
// ---------------------------------------------------------------------------
__global__ __launch_bounds__(256) void gemm_bf16_kernel(const bf16_t* A, const bf16_t* Bt,
                                                        const float* resid, float* C,
                                                        int M, int N, int K) {
  constexpr int LDT = 40;                            // 32 + 8 pad (80B rows, b128-clean)
  __shared__ alignas(16) bf16_t sA[128 * LDT];
  __shared__ alignas(16) bf16_t sB[128 * LDT];
  const int tid = threadIdx.x, lane = tid & 31, w = tid >> 5;
  const int lhi = lane >> 4, llo = lane & 15;
  const int m0 = blockIdx.x * 128, n0 = blockIdx.y * 128;
  const int lrow = tid >> 1, lhalf = tid & 1;

  v8f acc[8];
  for (int i = 0; i < 8; ++i) acc[i] = vzero8();

  const bf16_t* gA = A  + (size_t)(m0 + lrow) * K + lhalf * 16;
  const bf16_t* gB = Bt + (size_t)(n0 + lrow) * K + lhalf * 16;
  bf16_t* lA = sA + lrow * LDT + lhalf * 16;
  bf16_t* lB = sB + lrow * LDT + lhalf * 16;

  for (int k0 = 0; k0 < K; k0 += 32) {
    __syncthreads();                                 // prev-iter LDS consumers done
    async_copy_b128(gA,     lA);
    async_copy_b128(gA + 8, lA + 8);
    async_copy_b128(gB,     lB);
    async_copy_b128(gB + 8, lB + 8);
    if (k0 + 32 < K) {                               // near-temporal prefetch of next slab
      __builtin_prefetch(gA + 32, 0, 3);
      __builtin_prefetch(gB + 32, 0, 3);
    }
    wait_async0();                                   // s_wait_asynccnt 0
    __syncthreads();

    v16bf bfr[8];
    for (int n16 = 0; n16 < 8; ++n16)
      bfr[n16] = ld_frag(sB + (n16 * 16 + llo) * LDT + lhi * 8);
    v16bf af = ld_frag(sA + (w * 16 + llo) * LDT + lhi * 8);
    for (int n16 = 0; n16 < 8; ++n16)
      acc[n16] = wmma_bf16(af, bfr[n16], acc[n16]);

    gA += 32;
    gB += 32;
  }

  for (int n16 = 0; n16 < 8; ++n16) {
    for (int r = 0; r < 8; ++r) {
      const int row = m0 + w * 16 + r + lhi * 8;     // C layout: VGPR r -> M=r(+8)
      const int col = n0 + n16 * 16 + llo;           // N = lane&15
      const size_t idx = (size_t)row * N + col;
      float v = acc[n16][r];
      if (resid) v += resid[idx];
      C[idx] = v;
    }
  }
}

// ---------------------------------------------------------------------------
// Fused retention: per (b,h,64-row tile): ret = (q k^T * Dmat) v + (q h) * idec
// Output written in (b, s, h, d) order (ready for group-norm).
// ---------------------------------------------------------------------------
__global__ __launch_bounds__(256) void retention_kernel(const bf16_t* qg, const bf16_t* kg,
                                                        const bf16_t* vg, const bf16_t* hb,
                                                        const int* step, const int* dcx,
                                                        float* ret) {
  constexpr int LQ = 136;                            // 128 + 8 pad (272B rows)
  constexpr int LS = 72;                             // 64 + 8 pad (144B rows)
  __shared__ alignas(16) bf16_t qt[64 * LQ];
  __shared__ alignas(16) bf16_t kt[64 * LQ];
  __shared__ alignas(16) bf16_t scT[64 * LS];
  __shared__ alignas(16) bf16_t vT[128 * LS];
  __shared__ int tsn[64], dxn[64], tsm[64], dxm[64];

  const int bh = blockIdx.x, b = bh / H, h = bh % H;
  const int ntile = blockIdx.y, n0 = ntile * 64;
  const int tid = threadIdx.x, lane = tid & 31, w = tid >> 5;
  const int lhi = lane >> 4, llo = lane & 15;
  const float logk = __logf(1.f - exp2f(-5.f - (float)h));
  const int ts0 = step[b * S];

  // stage q tile (64 x 128) into LDS via the async DMA path
  {
    const int row = tid >> 2, seg = tid & 3;
    const bf16_t* src = qg + ((size_t)(b * S + n0 + row)) * D + h * HD + seg * 32;
    bf16_t* dst = qt + row * LQ + seg * 32;
    for (int j = 0; j < 4; ++j)
      async_copy_b128(src + j * 8, dst + j * 8);
  }
  if (tid < 64) {
    tsn[tid] = step[b * S + n0 + tid];
    dxn[tid] = dcx[b * S + n0 + tid];
  }
  wait_async0();
  __syncthreads();

  const int n16 = w >> 1;                            // query 16-row strip
  const int c0  = (w & 1) * 4;                       // output column-tile group
  const int mA  = (w & 1) * 2;                       // qk^T m16 pair

  v8f accO[4];
  for (int j = 0; j < 4; ++j) accO[j] = vzero8();

  for (int mt = 0; mt <= ntile; ++mt) {
    const int m0 = mt * 64;
    const int row = tid >> 2, seg = tid & 3;
    uint4 vreg[4];
    {
      const bf16_t* vsrc = vg + ((size_t)(b * S + m0 + row)) * D + h * HD + seg * 32;
      for (int j = 0; j < 4; ++j)
        vreg[j] = *(const uint4*)(vsrc + j * 8);
    }
    int tsv = 0, dxv = 0;
    if (tid < 64) {
      tsv = step[b * S + m0 + tid];
      dxv = dcx[b * S + m0 + tid];
    }
    __syncthreads();                                 // prev-iter LDS consumers done
    {
      const bf16_t* ksrc = kg + ((size_t)(b * S + m0 + row)) * D + h * HD + seg * 32;
      bf16_t* kdst = kt + row * LQ + seg * 32;
      for (int j = 0; j < 4; ++j)                    // k tile: straight async copy
        async_copy_b128(ksrc + j * 8, kdst + j * 8);
      union { uint4 u; bf16_t s[8]; } V;
      for (int j = 0; j < 4; ++j) {                  // transpose v -> vT[d][m]
        V.u = vreg[j];
        for (int q = 0; q < 8; ++q)
          vT[(seg * 32 + j * 8 + q) * LS + row] = V.s[q];
      }
      if (tid < 64) { tsm[tid] = tsv; dxm[tid] = dxv; }
    }
    wait_async0();
    __syncthreads();

    // Sc = q k^T  (K = 128 -> 4 wmma steps), then decay mask -> scT (bf16)
    v8f acc2[2];
    acc2[0] = vzero8();
    acc2[1] = vzero8();
    for (int ks = 0; ks < 4; ++ks) {
      v16bf bfr0 = ld_frag(kt + ((mA + 0) * 16 + llo) * LQ + ks * 32 + lhi * 8);
      v16bf bfr1 = ld_frag(kt + ((mA + 1) * 16 + llo) * LQ + ks * 32 + lhi * 8);
      v16bf af = ld_frag(qt + (n16 * 16 + llo) * LQ + ks * 32 + lhi * 8);
      acc2[0] = wmma_bf16(af, bfr0, acc2[0]);
      acc2[1] = wmma_bf16(af, bfr1, acc2[1]);
    }
    for (int j = 0; j < 2; ++j) {
      const int mloc = (mA + j) * 16 + llo;
      const int tm = tsm[mloc], dxmv = dxm[mloc];
      for (int r = 0; r < 8; ++r) {
        const int nloc = n16 * 16 + r + lhi * 8;
        const int tn = tsn[nloc];
        float dv = ((tn >= tm) && (dxn[nloc] == dxmv))
                       ? __expf((float)(tn - tm) * logk) : 0.f;
        scT[nloc * LS + mloc] = (bf16_t)(acc2[j][r] * dv);
      }
    }
    __syncthreads();

    // ret += Sc @ v   (K = 64 -> 2 wmma steps)
    for (int ks = 0; ks < 2; ++ks) {
      v16bf bfr[4];
      for (int j = 0; j < 4; ++j)
        bfr[j] = ld_frag(vT + ((c0 + j) * 16 + llo) * LS + ks * 32 + lhi * 8);
      v16bf af = ld_frag(scT + (n16 * 16 + llo) * LS + ks * 32 + lhi * 8);
      for (int j = 0; j < 4; ++j)
        accO[j] = wmma_bf16(af, bfr[j], accO[j]);
    }
  }
  __syncthreads();

  // cross-chunk: q @ hstate, scaled by inner_decay
  v8f accH[4];
  for (int j = 0; j < 4; ++j) accH[j] = vzero8();
  const bf16_t* hbb = hb + (size_t)bh * HD * HD;     // hb is [e][d] (transposed)
  for (int ks = 0; ks < 4; ++ks) {
    v16bf bfr[4];
    for (int j = 0; j < 4; ++j)
      bfr[j] = ld_frag(hbb + (size_t)((c0 + j) * 16 + llo) * HD + ks * 32 + lhi * 8);
    v16bf af = ld_frag(qt + (n16 * 16 + llo) * LQ + ks * 32 + lhi * 8);
    for (int j = 0; j < 4; ++j)
      accH[j] = wmma_bf16(af, bfr[j], accH[j]);
  }
  for (int j = 0; j < 4; ++j) {
    const int e = (c0 + j) * 16 + llo;
    for (int r = 0; r < 8; ++r) {
      const int nloc = n16 * 16 + r + lhi * 8;
      const float idec = (dxn[nloc] == 0)
                             ? __expf((float)(tsn[nloc] - ts0 + 1) * logk) : 0.f;
      const int n = n0 + nloc;
      ret[((size_t)(b * S + n)) * D + h * HD + e] = accO[j][r] + accH[j][r] * idec;
    }
  }
}

// ---------------------------------------------------------------------------
// next_h: per (b,h): 128x128 = (k*eta)^T @ v over S=512, + decayed carry.
// ---------------------------------------------------------------------------
__global__ __launch_bounds__(256) void next_h_kernel(const bf16_t* kg, const bf16_t* vg,
                                                     const float* hstate, const int* step,
                                                     const int* dcx, const unsigned char* dones,
                                                     float* outH) {
  constexpr int LS = 72;
  __shared__ alignas(16) bf16_t kT[128 * LS];        // [d][m]
  __shared__ alignas(16) bf16_t vT[128 * LS];        // [e][m]
  const int bh = blockIdx.x, b = bh / H, h = bh % H;
  const int tid = threadIdx.x, lane = tid & 31, w = tid >> 5;
  const int lhi = lane >> 4, llo = lane & 15;
  const float logk = __logf(1.f - exp2f(-5.f - (float)h));
  const int tsL = step[b * S + S - 1], ts0 = step[b * S];
  const int dcL = dcx[b * S + S - 1] + (dones[b * S + S - 1] ? 1 : 0);

  v8f acc[8];
  for (int j = 0; j < 8; ++j) acc[j] = vzero8();

  const int row = tid >> 2, seg = tid & 3;
  for (int mt = 0; mt < S / 64; ++mt) {
    const int m0 = mt * 64;
    uint4 kreg[4], vreg[4];
    const bf16_t* ksrc = kg + ((size_t)(b * S + m0 + row)) * D + h * HD + seg * 32;
    const bf16_t* vsrc = vg + ((size_t)(b * S + m0 + row)) * D + h * HD + seg * 32;
    for (int j = 0; j < 4; ++j) {
      kreg[j] = *(const uint4*)(ksrc + j * 8);
      vreg[j] = *(const uint4*)(vsrc + j * 8);
    }
    const int sm = b * S + m0 + row;
    const int dcm = dcx[sm] + (dones[sm] ? 1 : 0);
    const float eta = (dcm == dcL) ? __expf((float)(tsL - step[sm]) * logk) : 0.f;
    __syncthreads();
    union { uint4 u; bf16_t s[8]; } K8, V8;
    for (int j = 0; j < 4; ++j) {
      K8.u = kreg[j];
      V8.u = vreg[j];
      for (int q = 0; q < 8; ++q) {
        const int c = seg * 32 + j * 8 + q;
        kT[c * LS + row] = (bf16_t)((float)K8.s[q] * eta);
        vT[c * LS + row] = V8.s[q];
      }
    }
    __syncthreads();
    for (int ks = 0; ks < 2; ++ks) {
      v16bf af = ld_frag(kT + (w * 16 + llo) * LS + ks * 32 + lhi * 8);
      for (int j = 0; j < 8; j += 4) {
        v16bf bfr[4];
        for (int t = 0; t < 4; ++t)
          bfr[t] = ld_frag(vT + ((j + t) * 16 + llo) * LS + ks * 32 + lhi * 8);
        for (int t = 0; t < 4; ++t)
          acc[j + t] = wmma_bf16(af, bfr[t], acc[j + t]);
      }
    }
  }

  const float cf = (dcL == 0) ? __expf((float)(tsL - ts0 + 1) * logk) : 0.f;
  for (int j = 0; j < 8; ++j) {
    const int e = j * 16 + llo;
    for (int r = 0; r < 8; ++r) {
      const int dloc = w * 16 + r + lhi * 8;
      const size_t idx = (((size_t)bh) * HD + dloc) * HD + e;
      outH[idx] = acc[j][r] + cf * hstate[idx];
    }
  }
}

// ---------------------------------------------------------------------------
// Orchestration
// ---------------------------------------------------------------------------
extern "C" void kernel_launch(void* const* d_in, const int* in_sizes, int n_in,
                              void* d_out, int out_size, void* d_ws, size_t ws_size,
                              hipStream_t stream) {
  (void)in_sizes; (void)n_in; (void)out_size; (void)ws_size;

  const float* x        = (const float*)d_in[0];
  const float* hstate   = (const float*)d_in[1];
  const unsigned char* dones = (const unsigned char*)d_in[2];
  const int*   step     = (const int*)d_in[3];
  const float* ln1_w    = (const float*)d_in[4];
  const float* w_q      = (const float*)d_in[5];
  const float* w_k      = (const float*)d_in[6];
  const float* w_v      = (const float*)d_in[7];
  const float* w_g      = (const float*)d_in[8];
  const float* w_o      = (const float*)d_in[9];
  const float* gn_scale = (const float*)d_in[10];
  const float* gn_bias  = (const float*)d_in[11];
  const float* ln2_w    = (const float*)d_in[12];
  const float* W_gate   = (const float*)d_in[13];
  const float* W_linear = (const float*)d_in[14];
  const float* W_output = (const float*)d_in[15];
  float* out = (float*)d_out;

  const size_t DD  = (size_t)D * D;
  const size_t BSD = (size_t)BS * D;
  const size_t HH  = (size_t)B * H * HD * HD;

  char* ws = (char*)d_ws;
  size_t off = 0;
  auto alloc = [&](size_t bytes) -> char* {
    char* p = ws + off;
    off += (bytes + 255) & ~(size_t)255;
    return p;
  };

  const float* wsrc[8] = {w_q, w_k, w_v, w_g, w_o, W_gate, W_linear, W_output};
  bf16_t* wT[8];
  for (int i = 0; i < 8; ++i) wT[i] = (bf16_t*)alloc(DD * 2);
  float*  nxf  = (float*)alloc(BSD * 4);
  bf16_t* nxb  = (bf16_t*)alloc(BSD * 2);
  float*  qf   = (float*)alloc(BSD * 4);
  float*  kf   = (float*)alloc(BSD * 4);
  float*  vf   = (float*)alloc(BSD * 4);
  bf16_t* qb   = (bf16_t*)alloc(BSD * 2);
  bf16_t* kb   = (bf16_t*)alloc(BSD * 2);
  bf16_t* vb   = (bf16_t*)alloc(BSD * 2);
  bf16_t* hb   = (bf16_t*)alloc(HH * 2);
  float*  retb = (float*)alloc(BSD * 4);
  int*    dcx  = (int*)alloc((size_t)BS * 4);
  // aliases (lifetimes are disjoint in stream order)
  float*  gf    = qf;      // nx @ w_g
  float*  x2    = kf;      // x + attn
  float*  gatef = retb;    // n2 @ W_gate
  float*  linf  = vf;      // n2 @ W_linear
  bf16_t* gated = qb;      // silu(g) * rn
  bf16_t* n2b   = nxb;     // rmsnorm(x2)
  bf16_t* hmid  = vb;      // silu(gate) * lin

  const dim3 gg(BS / 128, D / 128);                  // 32 x 8 blocks

  // 1) precompute: transposed bf16 weights, transposed bf16 hstate, done-scan
  for (int i = 0; i < 8; ++i)
    transpose_w_kernel<<<(int)(DD / 256), 256, 0, stream>>>(wsrc[i], wT[i]);
  hb_transpose_kernel<<<(int)(HH / 256), 256, 0, stream>>>(hstate, hb);
  dcx_kernel<<<B, 1, 0, stream>>>(dones, dcx);

  // 2) nx = rmsnorm(x)
  rmsnorm_kernel<<<BS, 256, 0, stream>>>(x, ln1_w, nxf, nxb);

  // 3) q/k/v projections (WMMA GEMMs)
  gemm_bf16_kernel<<<gg, 256, 0, stream>>>(nxb, wT[0], nullptr, qf, BS, D, D);
  gemm_bf16_kernel<<<gg, 256, 0, stream>>>(nxb, wT[1], nullptr, kf, BS, D, D);
  gemm_bf16_kernel<<<gg, 256, 0, stream>>>(nxb, wT[2], nullptr, vf, BS, D, D);

  // 4) rotary (+ k scale) + bf16 cast
  const int rothreads = B * S * H * HALF;
  rotary_kernel<<<rothreads / 256, 256, 0, stream>>>(qf, qb, step, 1.0f);
  rotary_kernel<<<rothreads / 256, 256, 0, stream>>>(kf, kb, step, 0.08838834764831845f);
  cvt_f2b_kernel<<<(int)(BSD / 256), 256, 0, stream>>>(vf, vb, (int)BSD);

  // 5) fused retention + cross-chunk state
  retention_kernel<<<dim3(B * H, S / 64), 256, 0, stream>>>(qb, kb, vb, hb, step, dcx, retb);
  next_h_kernel<<<B * H, 256, 0, stream>>>(kb, vb, hstate, step, dcx, dones, out + BSD);

  // 6) gate projection, group-norm + SiLU gating, output projection (+x residual)
  gemm_bf16_kernel<<<gg, 256, 0, stream>>>(nxb, wT[3], nullptr, gf, BS, D, D);
  gn_gate_kernel<<<(B * S * H) / 8, 256, 0, stream>>>(retb, gf, gn_scale, gn_bias, gated);
  gemm_bf16_kernel<<<gg, 256, 0, stream>>>(gated, wT[4], x, x2, BS, D, D);

  // 7) FFN with residual fused into final GEMM
  rmsnorm_kernel<<<BS, 256, 0, stream>>>(x2, ln2_w, nullptr, n2b);
  gemm_bf16_kernel<<<gg, 256, 0, stream>>>(n2b, wT[5], nullptr, gatef, BS, D, D);
  gemm_bf16_kernel<<<gg, 256, 0, stream>>>(n2b, wT[6], nullptr, linf, BS, D, D);
  silu_mul_kernel<<<(int)(BSD / 256), 256, 0, stream>>>(gatef, linf, hmid, (int)BSD);
  gemm_bf16_kernel<<<gg, 256, 0, stream>>>(hmid, wT[7], x2, out, BS, D, D);
}
